// GNN_74019466379446
// MI455X (gfx1250) — matrix operand order repaired and checked by reading, not compiled, not run on hardware
//
#include <hip/hip_runtime.h>

// ---------------------------------------------------------------------------
// CDNA5 (gfx1250, wave32) fused GNN layers using v_wmma_f32_16x16x32_bf16.
// Fully templated on tile/shape constants so the A-operand gather is
// branch-free vectorized float4 loads and the K loop is compile-time.
// ---------------------------------------------------------------------------

typedef __attribute__((ext_vector_type(16))) __bf16 v16bf;
typedef __attribute__((ext_vector_type(8)))  float  v8f;

#define WAVES_PER_BLOCK 8
#define BLOCK_THREADS   (WAVES_PER_BLOCK * 32)

// ---------------------------------------------------------------------------
// Pack a row-major f32 weight matrix W[K x Hreal] into the wave32 WMMA
// B-operand layout, bf16, zero-padded to Kp (mult of 32) x NT*16 columns.
//   out[((kc*NT + t)*32 + lane)*16 + j]
//   lane 0-15  -> column t*16 + lane,     K = kc*32 + (j<8 ? j : 16+j-8)
//   lane 16-31 -> column t*16 + lane-16,  K = kc*32 + 8 + (j<8 ? j : 16+j-8)
// ---------------------------------------------------------------------------
__global__ void pack_weights_kernel(const float* __restrict__ W, int K, int Hreal,
                                    int Kp, int NT, __bf16* __restrict__ out) {
    int idx   = blockIdx.x * blockDim.x + threadIdx.x;
    int total = (Kp >> 5) * NT * 32;
    if (idx >= total) return;
    int lane = idx & 31;
    int t    = (idx >> 5) % NT;
    int kc   = idx / (32 * NT);
    int half = lane >> 4;
    int col  = t * 16 + (lane & 15);
    #pragma unroll
    for (int j = 0; j < 16; ++j) {
        int k = (kc << 5) + half * 8 + (j & 7) + ((j >> 3) << 4);
        float v = (k < K && col < Hreal) ? W[(size_t)k * Hreal + col] : 0.0f;
        out[((size_t)idx << 4) + j] = (__bf16)v;
    }
}

__global__ void fill_zero_kernel(float* __restrict__ p, unsigned long long n) {
    unsigned long long i = (unsigned long long)blockIdx.x * blockDim.x + threadIdx.x;
    if (i < n) p[i] = 0.0f;
}

// ---------------------------------------------------------------------------
// Load one run of 8 consecutive logical features starting at fs (fs % 8 == 0)
// from the virtual row [rowA(LENA) | rowB(LENB) | zeros]. LENA % 8 == 0, so a
// run is entirely in A, entirely in B(+pad), or entirely pad — never split
// across A/B. All region tests fold at compile time except one boundary
// select on `fs` (which only depends on kc/half).
// ---------------------------------------------------------------------------
template <int LENA, int LENB>
__device__ __forceinline__ void load_run8(const float* __restrict__ rowA,
                                          const float* __restrict__ rowB,
                                          int fs, float* __restrict__ r) {
    if (fs + 8 <= LENA) {                       // fully inside A
        float4 u = *(const float4*)(rowA + fs);
        float4 v = *(const float4*)(rowA + fs + 4);
        r[0]=u.x; r[1]=u.y; r[2]=u.z; r[3]=u.w;
        r[4]=v.x; r[5]=v.y; r[6]=v.z; r[7]=v.w;
    } else {                                     // fs >= LENA (both mult of 8)
        int fb = fs - LENA;
        if ((LENB & 7) == 0) {                   // B region itself 8-aligned
            if (fb + 8 <= LENB) {
                float4 u = *(const float4*)(rowB + fb);
                float4 v = *(const float4*)(rowB + fb + 4);
                r[0]=u.x; r[1]=u.y; r[2]=u.z; r[3]=u.w;
                r[4]=v.x; r[5]=v.y; r[6]=v.z; r[7]=v.w;
            } else {
                #pragma unroll
                for (int j = 0; j < 8; ++j) r[j] = 0.0f;
            }
        } else {                                 // LENB == 4 (edge_attr)
            if (fb == 0) {
                float4 u = *(const float4*)(rowB);
                r[0]=u.x; r[1]=u.y; r[2]=u.z; r[3]=u.w;
                r[4]=0.f; r[5]=0.f; r[6]=0.f; r[7]=0.f;
            } else {
                #pragma unroll
                for (int j = 0; j < 8; ++j) r[j] = 0.0f;
            }
        }
    }
}

// A-operand tile per ISA 16-bit A layout (16x32): lane L holds row M=L&15,
// half = L>>4; element j<8 -> K = k0 + half*8 + j ; j>=8 -> K = k0+16+half*8+(j-8).
template <int LENA, int LENB>
__device__ __forceinline__ v16bf gather_a_tile(const float* __restrict__ rowA,
                                               const float* __restrict__ rowB,
                                               int fs0 /* = kc*32 + half*8 */) {
    float r[16];
    load_run8<LENA, LENB>(rowA, rowB, fs0,      r);
    load_run8<LENA, LENB>(rowA, rowB, fs0 + 16, r + 8);
    v16bf a;
    #pragma unroll
    for (int j = 0; j < 16; ++j) a[j] = (__bf16)r[j];
    return a;
}

// ---------------------------------------------------------------------------
// Fused edge MLP + scatter-add:
//   m = relu([x[src[e]], edge_attr[e]] @ We + be);  aggr[dst[e]] += m
// One wave = 16 edges x NT*16 output columns; NK K-chunks of 32.
// ---------------------------------------------------------------------------
template <int NT, int NK, int LENA, int LENB, int HOUT>
__global__ void edge_msg_aggr_kernel(const float* __restrict__ xA, int strideA,
                                     const float* __restrict__ eattr,
                                     const long long* __restrict__ src,
                                     const long long* __restrict__ dst,
                                     const __bf16* __restrict__ Wp,
                                     const float* __restrict__ bias,
                                     int numTiles,
                                     float* __restrict__ aggr, int aggrStride) {
    int wave = threadIdx.x >> 5;
    int lane = threadIdx.x & 31;
    int tile = blockIdx.x * WAVES_PER_BLOCK + wave;
    if (tile >= numTiles) return;

    int row  = (tile << 4) | (lane & 15);
    int half = lane >> 4;
    long long s = src[row];
    const float* rowA = xA + (size_t)s * strideA;
    const float* rowB = eattr + (size_t)row * LENB;

    v8f zero = {};
    v8f acc[NT];
    #pragma unroll
    for (int t = 0; t < NT; ++t) acc[t] = zero;

    constexpr int CH = (NT >= 4) ? 4 : NT;
    #pragma unroll
    for (int kc = 0; kc < NK; ++kc) {
        v16bf a = gather_a_tile<LENA, LENB>(rowA, rowB, (kc << 5) + half * 8);
        if (kc + 1 < NK)
            __builtin_prefetch(Wp + ((((size_t)(kc + 1) * NT) * 32 + lane) << 4), 0, 3);
        #pragma unroll
        for (int tb = 0; tb < NT; tb += CH) {
            v16bf b[CH];
            #pragma unroll
            for (int u = 0; u < CH; ++u)
                b[u] = *(const v16bf*)(Wp + ((((size_t)kc * NT + tb + u) * 32 + lane) << 4));
            #pragma unroll
            for (int u = 0; u < CH; ++u)
                acc[tb + u] = __builtin_amdgcn_wmma_f32_16x16x32_bf16(
                    false, a, false, b[u], (short)0, acc[tb + u], false, false);
        }
    }

    // D layout: VGPR r, lanes 0-15 -> M=r ; lanes 16-31 -> M=8+r ; N = lane&15
    int colBase = lane & 15;
    float bcol[NT];
    #pragma unroll
    for (int t = 0; t < NT; ++t)
        bcol[t] = (t * 16 + colBase < HOUT) ? bias[t * 16 + colBase] : 0.0f;

    #pragma unroll
    for (int r = 0; r < 8; ++r) {
        int m = r + half * 8;
        long long d = dst[(tile << 4) + m];
        float* arow = aggr + (size_t)d * aggrStride;
        #pragma unroll
        for (int t = 0; t < NT; ++t) {
            int col = t * 16 + colBase;
            if (col < HOUT) {
                float v = acc[t][r] + bcol[t];
                v = v > 0.0f ? v : 0.0f;
                unsafeAtomicAdd(&arow[col], v);
            }
        }
    }
}

// ---------------------------------------------------------------------------
// Node MLP: h = relu([x, aggr] @ Wn + bn), one wave per 16-node tile.
// ---------------------------------------------------------------------------
template <int NT, int NK, int LENA, int LENB, int HOUT>
__global__ void node_update_kernel(const float* __restrict__ xA, int strideA,
                                   const float* __restrict__ xB,
                                   const __bf16* __restrict__ Wp,
                                   const float* __restrict__ bias,
                                   int numTiles,
                                   float* __restrict__ outp, int outStride) {
    int wave = threadIdx.x >> 5;
    int lane = threadIdx.x & 31;
    int tile = blockIdx.x * WAVES_PER_BLOCK + wave;
    if (tile >= numTiles) return;

    int node = (tile << 4) | (lane & 15);
    int half = lane >> 4;
    const float* rowA = xA + (size_t)node * strideA;
    const float* rowB = xB + (size_t)node * LENB;

    v8f zero = {};
    v8f acc[NT];
    #pragma unroll
    for (int t = 0; t < NT; ++t) acc[t] = zero;

    constexpr int CH = (NT >= 4) ? 4 : NT;
    #pragma unroll
    for (int kc = 0; kc < NK; ++kc) {
        v16bf a = gather_a_tile<LENA, LENB>(rowA, rowB, (kc << 5) + half * 8);
        if (kc + 1 < NK)
            __builtin_prefetch(Wp + ((((size_t)(kc + 1) * NT) * 32 + lane) << 4), 0, 3);
        #pragma unroll
        for (int tb = 0; tb < NT; tb += CH) {
            v16bf b[CH];
            #pragma unroll
            for (int u = 0; u < CH; ++u)
                b[u] = *(const v16bf*)(Wp + ((((size_t)kc * NT + tb + u) * 32 + lane) << 4));
            #pragma unroll
            for (int u = 0; u < CH; ++u)
                acc[tb + u] = __builtin_amdgcn_wmma_f32_16x16x32_bf16(
                    false, a, false, b[u], (short)0, acc[tb + u], false, false);
        }
    }

    int colBase = lane & 15;
    float bcol[NT];
    #pragma unroll
    for (int t = 0; t < NT; ++t)
        bcol[t] = (t * 16 + colBase < HOUT) ? bias[t * 16 + colBase] : 0.0f;

    #pragma unroll
    for (int r = 0; r < 8; ++r) {
        int m = r + half * 8;
        float* orow = outp + (size_t)((tile << 4) + m) * outStride;
        #pragma unroll
        for (int t = 0; t < NT; ++t) {
            int col = t * 16 + colBase;
            if (col < HOUT) {
                float v = acc[t][r] + bcol[t];
                orow[col] = v > 0.0f ? v : 0.0f;
            }
        }
    }
}

// ---------------------------------------------------------------------------
// Host-side orchestration
// ---------------------------------------------------------------------------
static inline int align32i(int k) { return ((k + 31) / 32) * 32; }

extern "C" void kernel_launch(void* const* d_in, const int* in_sizes, int n_in,
                              void* d_out, int out_size, void* d_ws, size_t ws_size,
                              hipStream_t stream) {
    const int H = 128, OUT = 8, NODE_IN = 8, EDGE_IN = 4;

    const float*     x   = (const float*)d_in[0];
    const long long* ei  = (const long long*)d_in[1];
    const float*     ea  = (const float*)d_in[2];
    const float*     We1 = (const float*)d_in[3];
    const float*     be1 = (const float*)d_in[4];
    const float*     Wn1 = (const float*)d_in[5];
    const float*     bn1 = (const float*)d_in[6];
    const float*     WeM = (const float*)d_in[7];
    const float*     beM = (const float*)d_in[8];
    const float*     WnM = (const float*)d_in[9];
    const float*     bnM = (const float*)d_in[10];
    const float*     WeC = (const float*)d_in[11];
    const float*     beC = (const float*)d_in[12];
    const float*     WnC = (const float*)d_in[13];
    const float*     bnC = (const float*)d_in[14];
    const float*     WeL = (const float*)d_in[15];
    const float*     beL = (const float*)d_in[16];
    const float*     WnL = (const float*)d_in[17];
    const float*     bnL = (const float*)d_in[18];

    const int N = in_sizes[0] / NODE_IN;   // 20000
    const int E = in_sizes[1] / 2;         // 320000
    const long long* src = ei;
    const long long* dst = ei + E;

    // ---- workspace carve-out -------------------------------------------
    unsigned char* wsb = (unsigned char*)d_ws;
    size_t off = 0;
    auto take = [&](size_t bytes) -> void* {
        void* p = wsb + off;
        off = (off + bytes + 255) & ~(size_t)255;
        return p;
    };

    float* aggr = (float*)take((size_t)N * H * sizeof(float));
    float* hcat = (float*)take((size_t)N * 7 * H * sizeof(float));   // h1..h7 column blocks
    float* h8   = (float*)take((size_t)N * H * sizeof(float));

    auto packElems = [&](int K, int NT) -> size_t {
        return (size_t)(align32i(K) / 32) * NT * 32 * 16;
    };
    __bf16* We1p = (__bf16*)take(packElems(12, 8) * 2);
    __bf16* Wn1p = (__bf16*)take(packElems(136, 8) * 2);
    __bf16* WeMp = (__bf16*)take(6 * packElems(132, 8) * 2);
    __bf16* WnMp = (__bf16*)take(6 * packElems(256, 8) * 2);
    __bf16* WeCp = (__bf16*)take(packElems(900, 8) * 2);
    __bf16* WnCp = (__bf16*)take(packElems(1024, 8) * 2);
    __bf16* WeLp = (__bf16*)take(packElems(132, 1) * 2);
    __bf16* WnLp = (__bf16*)take(packElems(136, 1) * 2);

    auto pack = [&](const float* W, int K, int Hreal, int NT, __bf16* outp) {
        int Kp = align32i(K);
        int total = (Kp / 32) * NT * 32;
        pack_weights_kernel<<<(total + 255) / 256, 256, 0, stream>>>(W, K, Hreal, Kp, NT, outp);
    };
    auto zero = [&](float* p, size_t n) {
        fill_zero_kernel<<<(unsigned)((n + 255) / 256), 256, 0, stream>>>(p, (unsigned long long)n);
    };

    const int edgeTiles  = E / 16;   // 20000, exact
    const int nodeTiles  = N / 16;   // 1250,  exact
    const int edgeBlocks = (edgeTiles + WAVES_PER_BLOCK - 1) / WAVES_PER_BLOCK;
    const int nodeBlocks = (nodeTiles + WAVES_PER_BLOCK - 1) / WAVES_PER_BLOCK;

    // ---- pack all weights (bf16, WMMA B layout) -------------------------
    pack(We1, 12, H, 8, We1p);
    pack(Wn1, 136, H, 8, Wn1p);
    for (int i = 0; i < 6; ++i) {
        pack(WeM + (size_t)i * 132 * H, 132, H, 8, WeMp + (size_t)i * packElems(132, 8));
        pack(WnM + (size_t)i * 256 * H, 256, H, 8, WnMp + (size_t)i * packElems(256, 8));
    }
    pack(WeC, 900, H, 8, WeCp);
    pack(WnC, 1024, H, 8, WnCp);
    pack(WeL, 132, OUT, 1, WeLp);
    pack(WnL, 136, OUT, 1, WnLp);

    // ---- layer 1 (K: 12 -> 32 ; 136 -> 160) -----------------------------
    zero(aggr, (size_t)N * H);
    edge_msg_aggr_kernel<8, 1, 8, 4, 128><<<edgeBlocks, BLOCK_THREADS, 0, stream>>>(
        x, NODE_IN, ea, src, dst, We1p, be1, edgeTiles, aggr, H);
    node_update_kernel<8, 5, 8, 128, 128><<<nodeBlocks, BLOCK_THREADS, 0, stream>>>(
        x, NODE_IN, aggr, Wn1p, bn1, nodeTiles, hcat, 7 * H);

    // ---- layers 2..7 (K: 132 -> 160 ; 256) ------------------------------
    for (int i = 0; i < 6; ++i) {
        zero(aggr, (size_t)N * H);
        edge_msg_aggr_kernel<8, 5, 128, 4, 128><<<edgeBlocks, BLOCK_THREADS, 0, stream>>>(
            hcat + i * H, 7 * H, ea, src, dst,
            WeMp + (size_t)i * packElems(132, 8), beM + i * H, edgeTiles, aggr, H);
        node_update_kernel<8, 8, 128, 128, 128><<<nodeBlocks, BLOCK_THREADS, 0, stream>>>(
            hcat + i * H, 7 * H, aggr,
            WnMp + (size_t)i * packElems(256, 8), bnM + i * H, nodeTiles,
            hcat + (i + 1) * H, 7 * H);
    }

    // ---- concat layer (K: 900 -> 928 ; 1024) ----------------------------
    zero(aggr, (size_t)N * H);
    edge_msg_aggr_kernel<8, 29, 896, 4, 128><<<edgeBlocks, BLOCK_THREADS, 0, stream>>>(
        hcat, 7 * H, ea, src, dst, WeCp, beC, edgeTiles, aggr, H);
    node_update_kernel<8, 32, 896, 128, 128><<<nodeBlocks, BLOCK_THREADS, 0, stream>>>(
        hcat, 7 * H, aggr, WnCp, bnC, nodeTiles, h8, H);

    // ---- last layer (OUT=8, NT=1, masked columns) -----------------------
    zero(aggr, (size_t)N * OUT);
    edge_msg_aggr_kernel<1, 5, 128, 4, 8><<<edgeBlocks, BLOCK_THREADS, 0, stream>>>(
        h8, H, ea, src, dst, WeLp, beL, edgeTiles, aggr, OUT);
    node_update_kernel<1, 5, 128, 8, 8><<<nodeBlocks, BLOCK_THREADS, 0, stream>>>(
        h8, H, aggr, WnLp, bnL, nodeTiles, (float*)d_out, OUT);

    (void)n_in; (void)out_size; (void)ws_size;
}
